// SDPLayer_71244917506211
// MI455X (gfx1250) — matrix operand order repaired and checked by reading, not compiled, not run on hardware
//
#include <hip/hip_runtime.h>

typedef __bf16 bf16;
typedef __attribute__((ext_vector_type(16))) __bf16 v16bf;
typedef __attribute__((ext_vector_type(8)))  float  v8f;

#define NMAX 1024
#define LDA  1024
#define ADMM_ITERS 50
#define NS_ITERS 16
#define NOUT 1000

// LDS tile: 64 rows x 32 bf16, row stride = 20 words (64B data + 16B pad).
// 16B-aligned fragment segments; matches TDM pad_interval=16dw / pad_amount=4dw.
#define ROWW 20

union Frag { v16bf v; unsigned int w[8]; };

__device__ __forceinline__ void split_f32(float x, bf16& h, bf16& l) {
  h = (bf16)x;
  l = (bf16)(x - (float)h);
}

#if defined(__has_builtin)
#if __has_builtin(__builtin_amdgcn_tensor_load_to_lds) && __has_builtin(__builtin_amdgcn_s_wait_tensorcnt)
#define USE_TDM 1
#endif
#endif

#ifdef USE_TDM
typedef __attribute__((ext_vector_type(4))) unsigned int v4u;
typedef __attribute__((ext_vector_type(8))) int v8i;
typedef __attribute__((ext_vector_type(4))) int v4i;

// DMA one 64x32 bf16 tile (row-major, 1024-wide source) into LDS at lds_off,
// with 16B padding inserted after every 64B row (=> 20-word LDS row stride).
__device__ __forceinline__ void tdm_load_tile(const bf16* gsrc, unsigned lds_off) {
  unsigned long long ga = (unsigned long long)(const void*)gsrc;
  v4u g0;
  g0[0] = 1u;                                        // count=1 (valid user D#)
  g0[1] = lds_off;                                   // lds_addr (bytes)
  g0[2] = (unsigned)(ga & 0xffffffffu);              // global_addr[31:0]
  g0[3] = (unsigned)((ga >> 32) & 0x01ffffffu)       // global_addr[56:32]
        | (2u << 30);                                // type = 2 ("image")
  v8i g1;
  g1[0] = (int)((1u << 16)       // data_size = 1 -> 2 bytes
              | (1u << 20)       // pad_enable
              | (3u << 22)       // pad_interval: code 3 -> every 16 DWORDs (64B)
              | (3u << 25));     // pad_amount:   code 3 -> 4 DWORDs (16B)
  g1[1] = (int)(1024u << 16);    // tensor_dim0 = 1024  (bits 79:48, low half)
  g1[2] = (int)(1024u << 16);    // tensor_dim1 = 1024  (bits 111:80, low half)
  g1[3] = (int)(32u << 16);      // tile_dim0 = 32      (bits 127:112)
  g1[4] = 64;                    // tile_dim1 = 64, tile_dim2 = 0
  g1[5] = 1024;                  // tensor_dim0_stride = 1024 (elements)
  g1[6] = 0;                     // stride0 hi / tensor_dim1_stride lo
  g1[7] = 0;
  v4i z4 = {0, 0, 0, 0};                  // groups 2/3 unused (2D tensor)
  v8i z8 = {0, 0, 0, 0, 0, 0, 0, 0};      // extra group (6-arg toolchain form)
  __builtin_amdgcn_tensor_load_to_lds(g0, g1, z4, z4, z8, 0);
}
#endif

// ---------------------------------------------------------------------------
// Split-bf16 GEMM:  D = alpha * (A x B) + beta * E, with A ~ Ah+Al, B ~ Bh+Bl.
// ALL operands in this solver are symmetric, so the B tile (K x N slab) is
// fetched as rows tileN..tileN+63 of B -- plain row-major 2D tiles for both
// operands, staged by the Tensor Data Mover.
// fp32 WMMA accumulation; epilogue optionally writes f32 D and its hi/lo split.
// Block = 128 threads (4 waves), 64x64 tile per block, 32x32 per wave.
// ---------------------------------------------------------------------------
__global__ __launch_bounds__(128)
void k_gemm_bf3(const bf16* __restrict__ Ah, const bf16* __restrict__ Al,
                const bf16* __restrict__ Bh, const bf16* __restrict__ Bl,
                const float* __restrict__ E, float alpha, float beta,
                float* __restrict__ D, bf16* __restrict__ Dh, bf16* __restrict__ Dl)
{
  __shared__ unsigned int sAh[64 * ROWW], sAl[64 * ROWW];
  __shared__ unsigned int sBh[64 * ROWW], sBl[64 * ROWW];

  const int tid   = threadIdx.x;
  const int lane  = tid & 31;
  const int wave  = tid >> 5;
  const int wr    = wave >> 1;       // wave row (0..1)
  const int wc    = wave & 1;        // wave col (0..1)
  const int tileM = blockIdx.y * 64;
  const int tileN = blockIdx.x * 64;
  const int lr    = lane & 15;       // lane within half-wave
  const int hs    = lane >> 4;       // half select

  v8f acc[2][2] = {};

  for (int k0 = 0; k0 < NMAX; k0 += 32) {
#ifdef USE_TDM
    if (wave == 0) {
      tdm_load_tile(Ah + (size_t)tileM * LDA + k0, (unsigned)(unsigned long long)(void*)sAh);
      tdm_load_tile(Al + (size_t)tileM * LDA + k0, (unsigned)(unsigned long long)(void*)sAl);
      tdm_load_tile(Bh + (size_t)tileN * LDA + k0, (unsigned)(unsigned long long)(void*)sBh);
      tdm_load_tile(Bl + (size_t)tileN * LDA + k0, (unsigned)(unsigned long long)(void*)sBl);
      __builtin_amdgcn_s_wait_tensorcnt(0);
    }
#else
    {
      bf16* aH = (bf16*)sAh; bf16* aL = (bf16*)sAl;
      bf16* bH = (bf16*)sBh; bf16* bL = (bf16*)sBl;
      #pragma unroll
      for (int it = 0; it < 16; ++it) {
        int e = it * 128 + tid;
        int r = e >> 5, c = e & 31;                   // tiles: 64 x 32, row-major
        aH[r * (2 * ROWW) + c] = Ah[(tileM + r) * LDA + (k0 + c)];
        aL[r * (2 * ROWW) + c] = Al[(tileM + r) * LDA + (k0 + c)];
        bH[r * (2 * ROWW) + c] = Bh[(tileN + r) * LDA + (k0 + c)];  // B symmetric
        bL[r * (2 * ROWW) + c] = Bl[(tileN + r) * LDA + (k0 + c)];
      }
    }
#endif
    __syncthreads();

    // ---- load fragments per ISA 16-bit A / B VGPR layouts ----
    Frag fAh[2], fAl[2], fBh[2], fBl[2];
    #pragma unroll
    for (int i = 0; i < 2; ++i) {
      const unsigned int* ph = &sAh[(wr * 32 + i * 16 + lr) * ROWW + hs * 4];
      const unsigned int* pl = &sAl[(wr * 32 + i * 16 + lr) * ROWW + hs * 4];
      #pragma unroll
      for (int q = 0; q < 4; ++q) {
        fAh[i].w[q] = ph[q]; fAh[i].w[4 + q] = ph[8 + q];
        fAl[i].w[q] = pl[q]; fAl[i].w[4 + q] = pl[8 + q];
      }
    }
    #pragma unroll
    for (int j = 0; j < 2; ++j) {
      const unsigned int* ph = &sBh[(wc * 32 + j * 16 + lr) * ROWW + hs * 4];
      const unsigned int* pl = &sBl[(wc * 32 + j * 16 + lr) * ROWW + hs * 4];
      #pragma unroll
      for (int q = 0; q < 4; ++q) {
        fBh[j].w[q] = ph[q]; fBh[j].w[4 + q] = ph[8 + q];
        fBl[j].w[q] = pl[q]; fBl[j].w[4 + q] = pl[8 + q];
      }
    }

    // ---- 3-product split-bf16 accumulation (drop Al*Bl term) ----
    #pragma unroll
    for (int i = 0; i < 2; ++i)
      #pragma unroll
      for (int j = 0; j < 2; ++j) {
        acc[i][j] = __builtin_amdgcn_wmma_f32_16x16x32_bf16(
            false, fAh[i].v, false, fBh[j].v, (short)0, acc[i][j], false, false);
        acc[i][j] = __builtin_amdgcn_wmma_f32_16x16x32_bf16(
            false, fAh[i].v, false, fBl[j].v, (short)0, acc[i][j], false, false);
        acc[i][j] = __builtin_amdgcn_wmma_f32_16x16x32_bf16(
            false, fAl[i].v, false, fBh[j].v, (short)0, acc[i][j], false, false);
      }
    __syncthreads();
  }

  // ---- epilogue: D = alpha*acc + beta*E, optional hi/lo split write ----
  #pragma unroll
  for (int i = 0; i < 2; ++i)
    #pragma unroll
    for (int j = 0; j < 2; ++j)
      #pragma unroll
      for (int v = 0; v < 8; ++v) {
        int row = tileM + wr * 32 + i * 16 + v + hs * 8;
        int col = tileN + wc * 32 + j * 16 + lr;
        float val = alpha * acc[i][j][v];
        if (E)  val += beta * E[row * LDA + col];
        if (D)  D[row * LDA + col] = val;
        if (Dh) { bf16 h, l; split_f32(val, h, l);
                  Dh[row * LDA + col] = h; Dl[row * LDA + col] = l; }
      }
}

// ---------------------------------------------------------------------------
// Elementwise / reduction helpers
// ---------------------------------------------------------------------------
__global__ void k_prep_w(const float* __restrict__ W, float* __restrict__ Wsym) {
  int idx = blockIdx.x * 256 + threadIdx.x;
  int r = idx >> 10, c = idx & 1023;
  Wsym[idx] = 0.5f * (W[idx] + W[c * LDA + r]);
}

__global__ void k_init_zu(float* __restrict__ Z, float* __restrict__ U) {
  int idx = blockIdx.x * 256 + threadIdx.x;
  int r = idx >> 10, c = idx & 1023;
  Z[idx] = (r == c) ? 1.0f : 0.0f;
  U[idx] = 0.0f;
}

__global__ void k_form_m(const float* __restrict__ Z, const float* __restrict__ U,
                         const float* __restrict__ Wsym, float* __restrict__ M) {
  int idx = blockIdx.x * 256 + threadIdx.x;
  M[idx] = Z[idx] - U[idx] + Wsym[idx];   // RHO == 1
}

__global__ void k_fnorm1(const float* __restrict__ M, float* __restrict__ part) {
  __shared__ float s[256];
  int t = threadIdx.x;
  int base = blockIdx.x * 1024;
  float acc = 0.0f;
  #pragma unroll
  for (int i = 0; i < 4; ++i) { float v = M[base + i * 256 + t]; acc += v * v; }
  s[t] = acc; __syncthreads();
  for (int o = 128; o > 0; o >>= 1) { if (t < o) s[t] += s[t + o]; __syncthreads(); }
  if (t == 0) part[blockIdx.x] = s[0];
}

__global__ void k_fnorm2(const float* __restrict__ part, float* __restrict__ invs) {
  __shared__ float s[256];
  int t = threadIdx.x;
  float acc = 0.0f;
  #pragma unroll
  for (int i = 0; i < 4; ++i) acc += part[i * 256 + t];
  s[t] = acc; __syncthreads();
  for (int o = 128; o > 0; o >>= 1) { if (t < o) s[t] += s[t + o]; __syncthreads(); }
  if (t == 0) { float f = s[0]; invs[0] = (f > 0.0f) ? rsqrtf(f) : 1.0f; }
}

// Y = M * inv_s (Newton-Schulz start, spectral norm <= 1), plus splits of Y and M
__global__ void k_split_scale(const float* __restrict__ M, const float* __restrict__ invs,
                              float* __restrict__ Y, bf16* __restrict__ Yh, bf16* __restrict__ Yl,
                              bf16* __restrict__ Mh, bf16* __restrict__ Ml) {
  int idx = blockIdx.x * 256 + threadIdx.x;
  float inv = invs[0];
  float m = M[idx];
  float y = m * inv;
  Y[idx] = y;
  bf16 h, l;
  split_f32(y, h, l); Yh[idx] = h; Yl[idx] = l;
  split_f32(m, h, l); Mh[idx] = h; Ml[idx] = l;
}

// Z = clip(X+U, 0) with unit diagonal;  U = (X+U) - Z
__global__ void k_zu(const float* __restrict__ X, float* __restrict__ Z, float* __restrict__ U) {
  int idx = blockIdx.x * 256 + threadIdx.x;
  int r = idx >> 10, c = idx & 1023;
  float t = X[idx] + U[idx];
  float z = fmaxf(t, 0.0f);
  if (r == c) z = 1.0f;
  Z[idx] = z;
  U[idx] = t - z;
}

__global__ void k_out(const float* __restrict__ Z, float* __restrict__ out) {
  int idx = blockIdx.x * 256 + threadIdx.x;
  if (idx < NOUT * NOUT) {
    int r = idx / NOUT, c = idx - r * NOUT;
    out[idx] = Z[r * LDA + c];
  }
}

// ---------------------------------------------------------------------------
extern "C" void kernel_launch(void* const* d_in, const int* in_sizes, int n_in,
                              void* d_out, int out_size, void* d_ws, size_t ws_size,
                              hipStream_t stream) {
  const float* W = (const float*)d_in[0];
  float* out = (float*)d_out;
  (void)in_sizes; (void)n_in; (void)out_size; (void)ws_size;

  const size_t NE = (size_t)NMAX * NMAX;
  char* p = (char*)d_ws;
  auto allocF = [&](size_t n) { float* r = (float*)p; p += ((n * 4 + 255) / 256) * 256; return r; };
  auto allocH = [&](size_t n) { bf16*  r = (bf16*)p;  p += ((n * 2 + 255) / 256) * 256; return r; };

  float* Wsym = allocF(NE); float* Z = allocF(NE); float* U = allocF(NE);
  float* M    = allocF(NE); float* X = allocF(NE);
  float* Ya   = allocF(NE); float* Yb = allocF(NE);
  bf16* Mh  = allocH(NE); bf16* Ml  = allocH(NE);
  bf16* Yah = allocH(NE); bf16* Yal = allocH(NE);
  bf16* Ybh = allocH(NE); bf16* Ybl = allocH(NE);
  bf16* Th  = allocH(NE); bf16* Tl  = allocH(NE);
  float* part = allocF(1024); float* invs = allocF(64);

  dim3 eb(256), eg((unsigned)(NE / 256));
  dim3 gb(128), gg(16, 16);

  k_prep_w<<<eg, eb, 0, stream>>>(W, Wsym);
  k_init_zu<<<eg, eb, 0, stream>>>(Z, U);

  for (int it = 0; it < ADMM_ITERS; ++it) {
    k_form_m<<<eg, eb, 0, stream>>>(Z, U, Wsym, M);
    k_fnorm1<<<dim3(1024), eb, 0, stream>>>(M, part);
    k_fnorm2<<<dim3(1), eb, 0, stream>>>(part, invs);
    k_split_scale<<<eg, eb, 0, stream>>>(M, invs, Ya, Yah, Yal, Mh, Ml);

    float* curF = Ya;  bf16* curH = Yah; bf16* curL = Yal;
    float* nxtF = Yb;  bf16* nxtH = Ybh; bf16* nxtL = Ybl;
    for (int ns = 0; ns < NS_ITERS; ++ns) {
      // T = Y * Y   (only the hi/lo split of T is needed)
      k_gemm_bf3<<<gg, gb, 0, stream>>>(curH, curL, curH, curL,
                                        nullptr, 1.0f, 0.0f,
                                        nullptr, Th, Tl);
      // Y' = 1.5*Y - 0.5*(Y*T)
      k_gemm_bf3<<<gg, gb, 0, stream>>>(curH, curL, Th, Tl,
                                        curF, -0.5f, 1.5f,
                                        nxtF, nxtH, nxtL);
      float* tf = curF; curF = nxtF; nxtF = tf;
      bf16* th = curH; curH = nxtH; nxtH = th;
      bf16* tl = curL; curL = nxtL; nxtL = tl;
    }

    // X = proj_PSD(M) = 0.5*(M + M*sign(M)),  sign(M) ~= Y_final
    k_gemm_bf3<<<gg, gb, 0, stream>>>(Mh, Ml, curH, curL,
                                      M, 0.5f, 0.5f,
                                      X, nullptr, nullptr);
    k_zu<<<eg, eb, 0, stream>>>(X, Z, U);
  }

  k_out<<<dim3((NOUT * NOUT + 255) / 256), eb, 0, stream>>>(Z, out);
}
